// UResNet_15350213116080
// MI455X (gfx1250) — matrix operand order repaired and checked by reading, not compiled, not run on hardware
//
#include <hip/hip_runtime.h>
#include <hip/hip_bf16.h>
#include <cstdint>
#include <cstddef>

// ---------------------------------------------------------------------------
// Sparse 3D U-Net (UResNet) forward on MI455X (gfx1250, wave32).
// Dense execution matching the reference; all conv/GEMM work runs on
// V_WMMA_F32_16X16X32_F16 (f16 inputs, fp32 accumulate). The 96^3 x 16ch fp32
// tensors (56.6MB) fit in the 192MB L2, so each level's working set stays
// L2-resident; conv inputs are staged channels-last f16 (half the bytes).
// Conv kernels are templated on (ks, KT, NT) with N-register-blocking:
// one wave = 16 voxels x full Cout, NT accumulators, each A fragment reused
// for all N tiles. Out-of-bounds conv taps read a zeroed guard row, so all
// inner-loop loads are unconditional (no EXEC masking / zero-fill VALU).
// ---------------------------------------------------------------------------

typedef _Float16 half_t;
typedef __attribute__((ext_vector_type(16))) half_t v16h;
typedef __attribute__((ext_vector_type(8)))  half_t v8h;
typedef __attribute__((ext_vector_type(8)))  float  v8f;

#define BN_EPS 1e-4f
#define MAXC   128

// ---------------------------- utility kernels ------------------------------

__global__ void k_scatter(const int* __restrict__ coords,
                          const float* __restrict__ feats,
                          float* __restrict__ mask0,
                          half_t* __restrict__ Hin,   // [nvox][32], pre-zeroed
                          int N, int D)
{
    int i = blockIdx.x * blockDim.x + threadIdx.x;
    if (i >= N) return;
    int x = coords[i * 3 + 0], y = coords[i * 3 + 1], z = coords[i * 3 + 2];
    long v = ((long)x * D + y) * D + z;
    mask0[v] = 1.0f;
    Hin[v * 32] = (half_t)feats[i];
}

__global__ void k_maxpool_mask(const float* __restrict__ mi,
                               float* __restrict__ mo, int Do)
{
    int idx = blockIdx.x * blockDim.x + threadIdx.x;
    int nvo = Do * Do * Do;
    if (idx >= nvo) return;
    int z = idx % Do, y = (idx / Do) % Do, x = idx / (Do * Do);
    int Di = Do * 2;
    float m = 0.f;
    for (int dx = 0; dx < 2; ++dx)
        for (int dy = 0; dy < 2; ++dy)
            for (int dz = 0; dz < 2; ++dz) {
                long vi = (((long)(2 * x + dx) * Di) + (2 * y + dy)) * Di + (2 * z + dz);
                m = fmaxf(m, mi[vi]);
            }
    mo[idx] = m;
}

__global__ void k_count(const float* __restrict__ mask, int nvox,
                        float* __restrict__ cnt)
{
    __shared__ float s;
    if (threadIdx.x == 0) s = 0.f;
    __syncthreads();
    float acc = 0.f;
    for (long i = (long)blockIdx.x * blockDim.x + threadIdx.x; i < nvox;
         i += (long)gridDim.x * blockDim.x)
        acc += mask[i];
    atomicAdd(&s, acc);               // ds_add_f32
    __syncthreads();
    if (threadIdx.x == 0) atomicAdd(cnt, s);
}

// per-channel sum / sumsq. blockDim.x == C (thread = channel), each block
// covers VPB voxels; register accumulation then two global atomics per thread.
#define VPB 128
__global__ void k_bn_stats(const float* __restrict__ X, int nvox,
                           float* __restrict__ st)
{
    const int C = blockDim.x;
    const int c = threadIdx.x;
    long v0 = (long)blockIdx.x * VPB;
    float s = 0.f, s2 = 0.f;
    for (int vv = 0; vv < VPB; ++vv) {
        long v = v0 + vv;
        if (v < nvox) {
            float x = X[v * C + c];
            s += x; s2 += x * x;
        }
    }
    atomicAdd(&st[c], s);
    atomicAdd(&st[C + c], s2);
}

// y = relu((x-mean)*rsqrt(var+eps)*g + b) * mask ;
// blockDim.x == Kpad (thread = channel), blockIdx.x == voxel.
__global__ void k_bn_apply(const float* __restrict__ X,
                           const float* __restrict__ mask,
                           const float* __restrict__ st,
                           const float* __restrict__ cnt,
                           const float* __restrict__ g,
                           const float* __restrict__ b,
                           int C,
                           half_t* __restrict__ H, float* __restrict__ Yf)
{
    const int Kpad = blockDim.x;
    const int c = threadIdx.x;
    const long v = blockIdx.x;
    float y = 0.f;
    if (c < C) {
        float n    = cnt[0];
        float mean = st[c] / n;
        float var  = st[C + c] / n - mean * mean;
        float xv   = X[v * C + c];
        y = (xv - mean) * rsqrtf(var + BN_EPS) * g[c] + b[c];
        y = fmaxf(y, 0.f) * mask[v];
        if (Yf) Yf[v * C + c] = y;
    }
    if (H) H[v * Kpad + c] = (half_t)y;
}

// fp32 [nvox][C] -> f16 [nvox][Kpad] (zero-padded K). blockDim.x == Kpad.
__global__ void k_stage(const float* __restrict__ X, half_t* __restrict__ H,
                        int C)
{
    const int Kpad = blockDim.x;
    const int c = threadIdx.x;
    const long v = blockIdx.x;
    H[v * Kpad + c] = (c < C) ? (half_t)X[v * C + c] : (half_t)0.f;
}

// blockDim.x == C, blockIdx.x == voxel
__global__ void k_copy_ch(const float* __restrict__ src, float* __restrict__ dst,
                          int dstStride, int dstOff)
{
    const int C = blockDim.x;
    const int c = threadIdx.x;
    const long v = blockIdx.x;
    dst[v * dstStride + dstOff + c] = src[v * C + c];
}

// Repack fp32 weights [taps][Cin][Cout] into WMMA B fragments (f16):
// layout [t][kt][nt][lane][16 halves]; half h -> K=(h&7)+(h>=8?16:0)+(lane>=16?8:0),
// N=lane&15. flip=1 reverses tap order (transposed conv).
__global__ void k_pack_w(const float* __restrict__ w, half_t* __restrict__ out,
                         int taps, int KT, int NT, int Cin, int Cout, int flip)
{
    long total = (long)taps * KT * NT * 512;
    for (long i = (long)blockIdx.x * blockDim.x + threadIdx.x; i < total;
         i += (long)gridDim.x * blockDim.x) {
        int  h = (int)(i & 15);
        int  L = (int)((i >> 4) & 31);
        long q = i >> 9;
        int  nt = (int)(q % NT); q /= NT;
        int  kt = (int)(q % KT);
        int  t  = (int)(q / KT);
        int  K = (h & 7) + ((h >= 8) ? 16 : 0) + ((L >= 16) ? 8 : 0);
        int  Nn = L & 15;
        int  ci = kt * 32 + K, co = nt * 16 + Nn;
        int  ts = flip ? (taps - 1 - t) : t;
        float v = (ci < Cin && co < Cout) ? w[((long)ts * Cin + ci) * Cout + co] : 0.f;
        out[i] = (half_t)v;
    }
}

// ------------------------- WMMA convolution kernels ------------------------
// A-fragment per-lane layout (ISA 16-bit A-matrix): lanes 0-15 K=[0..7,16..23],
// lanes 16-31 K=[8..15,24..31]; two aligned b128 loads per lane per K-tile.
// arow already points at the zero guard row for OOB lanes -> unconditional.

__device__ __forceinline__ v16h load_a(const half_t* arow, int kt, int koff)
{
    v8h lo  = *(const v8h*)(arow + kt * 32 + koff);
    v8h hi2 = *(const v8h*)(arow + kt * 32 + 16 + koff);
    v16h a;
#pragma unroll
    for (int e = 0; e < 8; ++e) { a[e] = lo[e]; a[e + 8] = hi2[e]; }
    return a;
}

// Templated direct conv: ks^3 shifted GEMMs, one wave = 16 voxels x NT*16 couts
// with NT register accumulators; each A fragment feeds NT WMMAs.
template <int KS, int KT, int NT>
__global__ void __launch_bounds__(256)
k_conv_t(const half_t* __restrict__ A,    // [nvoxIn][KT*32] f16
         const half_t* __restrict__ W,    // packed B fragments
         const half_t* __restrict__ Z,    // zero guard row (>= KT*32 halves)
         float* __restrict__ O,
         const float* __restrict__ mask,  // nullable, at output level
         const float* __restrict__ add,   // nullable residual [nvoxOut][Cout]
         int Din, int Dout, int stride, int pad)
{
    constexpr int TAPS = KS * KS * KS;
    constexpr int Kpad = KT * 32;
    constexpr int Cout = NT * 16;
    const int nvo = Dout * Dout * Dout;
    const int MT  = (nvo + 15) >> 4;
    const int wid = (int)(((long)blockIdx.x * blockDim.x + threadIdx.x) >> 5);
    if (wid >= MT) return;
    const int mtile = wid;
    const int lane  = threadIdx.x & 31;
    const int Mrow  = lane & 15;
    const int hiw   = lane >> 4;
    const int koff  = hiw << 3;
    const int vout  = mtile * 16 + Mrow;
    const bool rowValid = vout < nvo;
    int X = 0, Y = 0, Z3 = 0;
    if (rowValid) { Z3 = vout % Dout; int q = vout / Dout; Y = q % Dout; X = q / Dout; }

    v8f zero = {};
    v8f acc[NT];
#pragma unroll
    for (int nt = 0; nt < NT; ++nt) acc[nt] = zero;

#pragma unroll 1
    for (int t = 0; t < TAPS; ++t) {
        const int tz = t % KS, q = t / KS;
        const int ty = q % KS, tx = q / KS;
        const int nx = X * stride + tx - pad;
        const int ny = Y * stride + ty - pad;
        const int nz = Z3 * stride + tz - pad;
        const bool ok = rowValid && (unsigned)nx < (unsigned)Din &&
                        (unsigned)ny < (unsigned)Din && (unsigned)nz < (unsigned)Din;
        const half_t* arow = ok ? (A + (((long)nx * Din + ny) * Din + nz) * Kpad) : Z;
        const half_t* wt   = W + ((long)t * KT * NT) * 512 + lane * 16;
        if (t + 1 < TAPS)  // pull next tap's weight block toward the WGP
            __builtin_prefetch(wt + (long)KT * NT * 512, 0, 0);
#pragma unroll
        for (int kt = 0; kt < KT; ++kt) {
            v16h a = load_a(arow, kt, koff);
#pragma unroll
            for (int nt = 0; nt < NT; ++nt) {
                v16h b = *(const v16h*)(wt + ((long)kt * NT + nt) * 512);
                acc[nt] = __builtin_amdgcn_wmma_f32_16x16x32_f16(
                    false, a, false, b, (short)0, acc[nt], false, false);
            }
        }
    }
    const int mb = hiw ? 8 : 0;
#pragma unroll
    for (int nt = 0; nt < NT; ++nt) {
        const int co = nt * 16 + Mrow;
#pragma unroll
        for (int r = 0; r < 8; ++r) {
            const int v = mtile * 16 + mb + r;
            if (v < nvo) {
                float o = acc[nt][r];
                if (mask) o *= mask[v];
                if (add)  o += add[(long)v * Cout + co];
                O[(long)v * Cout + co] = o;
            }
        }
    }
}

// Templated transposed 2x2x2 stride-2 conv: out[2u+o] = in[u] . w[1-o]
// (tap flip at pack time). blockIdx.y = tap; one wave per 16-coarse-voxel tile.
template <int KT, int NT>
__global__ void __launch_bounds__(256)
k_upconv_t(const half_t* __restrict__ A,   // coarse [nvoxC][KT*32]
           const half_t* __restrict__ W,   // packed, 8 taps, pre-flipped
           const half_t* __restrict__ Z,   // zero guard row
           float* __restrict__ O,          // fine level
           const float* __restrict__ mask, // fine level
           int Dc, int oStride, int oOff)
{
    constexpr int Kpad = KT * 32;
    const int nvc = Dc * Dc * Dc;
    const int MT  = (nvc + 15) >> 4;
    const int wid = (int)(((long)blockIdx.x * blockDim.x + threadIdx.x) >> 5);
    if (wid >= MT) return;
    const int t     = blockIdx.y;    // tap 0..7
    const int mtile = wid;
    const int lane  = threadIdx.x & 31;
    const int Mrow  = lane & 15;
    const int hiw   = lane >> 4;
    const int koff  = hiw << 3;
    const int u     = mtile * 16 + Mrow;
    const bool ok   = u < nvc;
    const half_t* arow = ok ? (A + (long)u * Kpad) : Z;
    const half_t* wt   = W + ((long)t * KT * NT) * 512 + lane * 16;

    v8f zero = {};
    v8f acc[NT];
#pragma unroll
    for (int nt = 0; nt < NT; ++nt) acc[nt] = zero;
#pragma unroll
    for (int kt = 0; kt < KT; ++kt) {
        v16h a = load_a(arow, kt, koff);
#pragma unroll
        for (int nt = 0; nt < NT; ++nt) {
            v16h b = *(const v16h*)(wt + ((long)kt * NT + nt) * 512);
            acc[nt] = __builtin_amdgcn_wmma_f32_16x16x32_f16(
                false, a, false, b, (short)0, acc[nt], false, false);
        }
    }
    const int tz = t % 2, q2 = t / 2;
    const int ty = q2 % 2, tx = q2 / 2;
    const int Df = Dc * 2;
    const int mb = hiw ? 8 : 0;
#pragma unroll
    for (int nt = 0; nt < NT; ++nt) {
        const int co = nt * 16 + Mrow;
#pragma unroll
        for (int r = 0; r < 8; ++r) {
            const int uu = mtile * 16 + mb + r;
            if (uu < nvc) {
                int z = uu % Dc; int qq = uu / Dc; int y = qq % Dc; int x = qq / Dc;
                long v = (((long)(2 * x + tx) * Df) + (2 * y + ty)) * Df + (2 * z + tz);
                O[v * oStride + oOff + co] = acc[nt][r] * mask[v];
            }
        }
    }
}

// --------- generic runtime fallback (any shape; keeps dispatch total) -------

__global__ void __launch_bounds__(256)
k_conv_gen(const half_t* __restrict__ A, const half_t* __restrict__ W,
           const half_t* __restrict__ Z,
           float* __restrict__ O, const float* __restrict__ mask,
           const float* __restrict__ add,
           int Din, int Dout, int Kpad, int Cout,
           int ks, int stride, int pad)
{
    const int NT  = Cout >> 4;
    const int KT  = Kpad >> 5;
    const int nvo = Dout * Dout * Dout;
    const int MT  = (nvo + 15) >> 4;
    const int wid = (int)(((long)blockIdx.x * blockDim.x + threadIdx.x) >> 5);
    if (wid >= MT * NT) return;
    const int mtile = wid / NT;
    const int ntile = wid - mtile * NT;
    const int lane  = threadIdx.x & 31;
    const int Mrow  = lane & 15;
    const int hiw   = lane >> 4;
    const int koff  = hiw << 3;
    const int vout  = mtile * 16 + Mrow;
    const bool rowValid = vout < nvo;
    int X = 0, Y = 0, Z3 = 0;
    if (rowValid) { Z3 = vout % Dout; int q = vout / Dout; Y = q % Dout; X = q / Dout; }
    v8f acc = {};
    const int taps = ks * ks * ks;
    for (int t = 0; t < taps; ++t) {
        const int tz = t % ks, q = t / ks;
        const int ty = q % ks, tx = q / ks;
        const int nx = X * stride + tx - pad;
        const int ny = Y * stride + ty - pad;
        const int nz = Z3 * stride + tz - pad;
        const bool ok = rowValid && (unsigned)nx < (unsigned)Din &&
                        (unsigned)ny < (unsigned)Din && (unsigned)nz < (unsigned)Din;
        const half_t* arow = ok ? (A + (((long)nx * Din + ny) * Din + nz) * Kpad) : Z;
        const half_t* wrow = W + (((long)t * KT) * NT + ntile) * 512 + lane * 16;
        for (int kt = 0; kt < KT; ++kt) {
            v16h a = load_a(arow, kt, koff);
            v16h b = *(const v16h*)(wrow + (long)kt * NT * 512);
            acc = __builtin_amdgcn_wmma_f32_16x16x32_f16(
                false, a, false, b, (short)0, acc, false, false);
        }
    }
    const int co = ntile * 16 + Mrow;
    const int mb = hiw ? 8 : 0;
#pragma unroll
    for (int r = 0; r < 8; ++r) {
        const int v = mtile * 16 + mb + r;
        if (v < nvo) {
            float o = acc[r];
            if (mask) o *= mask[v];
            if (add)  o += add[(long)v * Cout + co];
            O[(long)v * Cout + co] = o;
        }
    }
}

__global__ void k_gather_linear(const float* __restrict__ X,  // [nvox][16]
                                const int* __restrict__ coords,
                                const float* __restrict__ lw,  // [16][5]
                                const float* __restrict__ lb,  // [5]
                                float* __restrict__ out, int N, int D)
{
    int i = blockIdx.x * blockDim.x + threadIdx.x;
    if (i >= N) return;
    int x = coords[i * 3], y = coords[i * 3 + 1], z = coords[i * 3 + 2];
    long v = ((long)x * D + y) * D + z;
    for (int c = 0; c < 5; ++c) {
        float s = lb[c];
        for (int k = 0; k < 16; ++k) s += X[v * 16 + k] * lw[k * 5 + c];
        out[i * 5 + c] = s;
    }
}

// ----------------------------- host dispatch --------------------------------

static inline dim3 gwaves(long waves) { return dim3((unsigned)((waves * 32 + 255) / 256)); }

static void launch_conv(hipStream_t s, const half_t* A, const half_t* W,
                        const half_t* Z, float* O,
                        const float* mask, const float* add,
                        int Din, int Dout, int Kpad, int Cout,
                        int ks, int strd, int pad)
{
    const int NT = Cout >> 4, KT = Kpad >> 5;
    const int nvo = Dout * Dout * Dout, MT = (nvo + 15) >> 4;
#define CONV_CASE(KS_, KT_, NT_)                                               \
    if (ks == KS_ && KT == KT_ && NT == NT_) {                                 \
        k_conv_t<KS_, KT_, NT_><<<gwaves(MT), 256, 0, s>>>(                    \
            A, W, Z, O, mask, add, Din, Dout, strd, pad);                      \
        return;                                                                \
    }
    CONV_CASE(3, 1, 1) CONV_CASE(3, 1, 2) CONV_CASE(3, 2, 2) CONV_CASE(3, 2, 3)
    CONV_CASE(3, 2, 4) CONV_CASE(3, 3, 3) CONV_CASE(3, 3, 5) CONV_CASE(3, 4, 4)
    CONV_CASE(2, 1, 2) CONV_CASE(2, 1, 3) CONV_CASE(2, 2, 4) CONV_CASE(2, 2, 5)
    CONV_CASE(1, 1, 1) CONV_CASE(1, 2, 2) CONV_CASE(1, 3, 3) CONV_CASE(1, 4, 4)
#undef CONV_CASE
    k_conv_gen<<<gwaves((long)MT * NT), 256, 0, s>>>(A, W, Z, O, mask, add, Din,
                                                     Dout, Kpad, Cout, ks, strd, pad);
}

static void launch_upconv(hipStream_t s, const half_t* A, const half_t* W,
                          const half_t* Z, float* O,
                          const float* mask, int Dc, int Kpad, int Cout,
                          int oStride, int oOff)
{
    const int NT = Cout >> 4, KT = Kpad >> 5;
    const int nvc = Dc * Dc * Dc, MT = (nvc + 15) >> 4;
    dim3 grid(gwaves(MT).x, 8);   // blockIdx.y = tap
#define UP_CASE(KT_, NT_)                                                      \
    if (KT == KT_ && NT == NT_) {                                              \
        k_upconv_t<KT_, NT_><<<grid, 256, 0, s>>>(A, W, Z, O, mask, Dc,        \
                                                  oStride, oOff);              \
        return;                                                                \
    }
    UP_CASE(1, 1) UP_CASE(2, 2) UP_CASE(2, 3) UP_CASE(3, 4)
#undef UP_CASE
    // all shapes in this net are covered above (shapes are fixed)
}

// ------------------------------- entry point --------------------------------

extern "C" void kernel_launch(void* const* d_in, const int* in_sizes, int n_in,
                              void* d_out, int out_size, void* d_ws, size_t ws_size,
                              hipStream_t stream)
{
    (void)in_sizes; (void)n_in; (void)out_size; (void)ws_size;
    static const int NPT = 120000;
    static const int Dl[5] = {96, 48, 24, 12, 6};
    static const int nv[5] = {884736, 110592, 13824, 1728, 216};
    static const int Cc[5] = {16, 32, 48, 64, 80};
    auto pad32 = [](int c) { return (c + 31) & ~31; };

    // ---- inputs in jax tree_flatten order (dict keys sorted) ----
    const int*   coords   = (const int*)d_in[0];
    const float* features = (const float*)d_in[1];
    int ip = 2;
    auto nxt = [&]() { return (const float*)d_in[ip++]; };
    struct BNp { const float* g; const float* b; };
    struct Blk { BNp bn1, bn2; const float* c1; const float* c2; const float* nin;
                 half_t* p1; half_t* p2; half_t* pnin; };

    BNp bn_out; bn_out.b = nxt(); bn_out.g = nxt();
    Blk dec[4][2];
    for (int i = 0; i < 4; ++i)
        for (int j = 0; j < 2; ++j) {
            Blk& B = dec[i][j];
            B.bn1.b = nxt(); B.bn1.g = nxt();
            B.bn2.b = nxt(); B.bn2.g = nxt();
            B.c1 = nxt(); B.c2 = nxt();
            B.nin = (j == 0) ? nxt() : nullptr;
        }
    BNp down_bn[4]; const float* down_w[4];
    for (int i = 0; i < 4; ++i) { down_bn[i].b = nxt(); down_bn[i].g = nxt(); down_w[i] = nxt(); }
    Blk enc[5][2];
    for (int l = 0; l < 5; ++l)
        for (int j = 0; j < 2; ++j) {
            Blk& B = enc[l][j];
            B.bn1.b = nxt(); B.bn1.g = nxt();
            B.bn2.b = nxt(); B.bn2.g = nxt();
            B.c1 = nxt(); B.c2 = nxt(); B.nin = nullptr;
        }
    const float* lin_b = nxt();
    const float* lin_w = nxt();
    BNp up_bn[4]; const float* up_w[4];
    for (int i = 0; i < 4; ++i) { up_bn[i].b = nxt(); up_bn[i].g = nxt(); up_w[i] = nxt(); }
    const float* w_in = nxt();

    // ---- workspace bump allocator ----
    char* wsp = (char*)d_ws; size_t off = 0;
    auto alloc = [&](size_t bytes) -> void* {
        void* p = wsp + off;
        off = (off + bytes + 255) & ~(size_t)255;
        return p;
    };
    float* mask[5]; for (int l = 0; l < 5; ++l) mask[l] = (float*)alloc((size_t)nv[l] * 4);
    float* ncnt  = (float*)alloc(5 * 4);
    float* stats = (float*)alloc(2 * MAXC * 4);
    half_t* Zrow = (half_t*)alloc(256 * 2);             // zero guard row
    float* xb[5]; for (int l = 0; l < 5; ++l) xb[l] = (float*)alloc((size_t)nv[l] * Cc[l] * 4);
    float* cat[4]; for (int l = 0; l < 4; ++l) cat[l] = (float*)alloc((size_t)nv[l] * 2 * Cc[l] * 4);
    float*  tmp1 = (float*)alloc((size_t)nv[0] * 16 * 4);
    float*  tmp2 = (float*)alloc((size_t)nv[0] * 16 * 4);
    half_t* Hbuf = (half_t*)alloc((size_t)nv[0] * 32 * 2);

    auto g1 = [](long total) { return dim3((unsigned)((total + 255) / 256)); };

    auto packW = [&](const float* w, int ks, int Cin, int Cout, int flip) -> half_t* {
        int taps = ks * ks * ks, KT = (Cin + 31) / 32, NT = Cout / 16;
        long elems = (long)taps * KT * NT * 512;
        half_t* p = (half_t*)alloc((size_t)elems * 2);
        k_pack_w<<<g1(elems), 256, 0, stream>>>(w, p, taps, KT, NT, Cin, Cout, flip);
        return p;
    };

    // ---- repack all weights into WMMA B-fragment layout ----
    half_t* win_p = packW(w_in, 3, 1, 16, 0);
    half_t* down_p[4]; half_t* up_p[4];
    for (int i = 0; i < 4; ++i) {
        down_p[i] = packW(down_w[i], 2, Cc[i], Cc[i + 1], 0);
        up_p[i]   = packW(up_w[i],   2, Cc[i + 1], Cc[i], 1);  // tap-flipped
    }
    for (int l = 0; l < 5; ++l)
        for (int j = 0; j < 2; ++j) {
            enc[l][j].p1 = packW(enc[l][j].c1, 3, Cc[l], Cc[l], 0);
            enc[l][j].p2 = packW(enc[l][j].c2, 3, Cc[l], Cc[l], 0);
        }
    for (int i = 0; i < 4; ++i) {
        dec[i][0].pnin = packW(dec[i][0].nin, 1, 2 * Cc[i], Cc[i], 0);
        dec[i][0].p1   = packW(dec[i][0].c1, 3, 2 * Cc[i], Cc[i], 0);
        dec[i][0].p2   = packW(dec[i][0].c2, 3, Cc[i], Cc[i], 0);
        dec[i][1].p1   = packW(dec[i][1].c1, 3, Cc[i], Cc[i], 0);
        dec[i][1].p2   = packW(dec[i][1].c2, 3, Cc[i], Cc[i], 0);
    }

    // ---- masks, counts, guard row, input scatter ----
    hipMemsetAsync(mask[0], 0, (size_t)nv[0] * 4, stream);
    hipMemsetAsync(ncnt, 0, 5 * 4, stream);
    hipMemsetAsync(Zrow, 0, 256 * 2, stream);
    hipMemsetAsync(Hbuf, 0, (size_t)nv[0] * 32 * 2, stream);
    k_scatter<<<g1(NPT), 256, 0, stream>>>(coords, features, mask[0], Hbuf, NPT, 96);
    for (int l = 1; l < 5; ++l)
        k_maxpool_mask<<<g1(nv[l]), 256, 0, stream>>>(mask[l - 1], mask[l], Dl[l]);
    for (int l = 0; l < 5; ++l)
        k_count<<<g1(nv[l]), 256, 0, stream>>>(mask[l], nv[l], ncnt + l);

    // ---- helpers ----
    auto conv = [&](const half_t* A, const half_t* W, float* O,
                    const float* msk, const float* add,
                    int Din, int Dout, int Kpad, int Cout,
                    int ks, int strd, int pd) {
        launch_conv(stream, A, W, Zrow, O, msk, add, Din, Dout, Kpad, Cout, ks, strd, pd);
    };
    auto bnrelu = [&](const float* X, int lvl, BNp bn, int nvox, int C, int Kpad,
                      half_t* H, float* Yf) {
        hipMemsetAsync(stats, 0, (size_t)2 * C * 4, stream);
        k_bn_stats<<<dim3((unsigned)((nvox + VPB - 1) / VPB)), C, 0, stream>>>(X, nvox, stats);
        k_bn_apply<<<dim3((unsigned)nvox), Kpad, 0, stream>>>(
            X, mask[lvl], stats, ncnt + lvl, bn.g, bn.b, C, H, Yf);
    };
    auto res_block = [&](int lvl, Blk& B, const float* Xin, int Cin, float* Xout) {
        int Cout = Cc[lvl], Dd = Dl[lvl], nvox = nv[lvl];
        int Kin = pad32(Cin), Kmid = pad32(Cout);
        const float* sc = Xin;
        if (B.nin) {  // shortcut projection sc = Xin @ nin (1x1x1 conv)
            k_stage<<<dim3((unsigned)nvox), Kin, 0, stream>>>(Xin, Hbuf, Cin);
            conv(Hbuf, B.pnin, tmp2, nullptr, nullptr, Dd, Dd, Kin, Cout, 1, 1, 0);
            sc = tmp2;
        }
        bnrelu(Xin, lvl, B.bn1, nvox, Cin, Kin, Hbuf, nullptr);
        conv(Hbuf, B.p1, tmp1, mask[lvl], nullptr, Dd, Dd, Kin, Cout, 3, 1, 1);
        bnrelu(tmp1, lvl, B.bn2, nvox, Cout, Kmid, Hbuf, nullptr);
        conv(Hbuf, B.p2, Xout, mask[lvl], sc, Dd, Dd, Kmid, Cout, 3, 1, 1);
    };

    // ---- forward ----
    // w_in: Hbuf holds scattered features (1 ch, padded to 32)
    conv(Hbuf, win_p, xb[0], mask[0], nullptr, 96, 96, 32, 16, 3, 1, 1);

    // encoder
    for (int l = 0; l < 5; ++l) {
        res_block(l, enc[l][0], xb[l], Cc[l], xb[l]);
        res_block(l, enc[l][1], xb[l], Cc[l], xb[l]);
        if (l < 4) {
            bnrelu(xb[l], l, down_bn[l], nv[l], Cc[l], pad32(Cc[l]), Hbuf, nullptr);
            conv(Hbuf, down_p[l], xb[l + 1], nullptr, nullptr,
                 Dl[l], Dl[l + 1], pad32(Cc[l]), Cc[l + 1], 2, 2, 0);
        }
    }
    // decoder
    for (int l = 3; l >= 0; --l) {
        int Cf = Cc[l + 1];
        bnrelu(xb[l + 1], l + 1, up_bn[l], nv[l + 1], Cf, pad32(Cf), Hbuf, nullptr);
        launch_upconv(stream, Hbuf, up_p[l], Zrow, cat[l], mask[l], Dl[l + 1],
                      pad32(Cf), Cc[l], 2 * Cc[l], Cc[l]);
        k_copy_ch<<<dim3((unsigned)nv[l]), Cc[l], 0, stream>>>(
            xb[l], cat[l], 2 * Cc[l], 0);
        res_block(l, dec[l][0], cat[l], 2 * Cc[l], xb[l]);
        res_block(l, dec[l][1], xb[l], Cc[l], xb[l]);
    }

    // bn_out (fp32) + gather + linear head
    bnrelu(xb[0], 0, bn_out, nv[0], 16, 16, nullptr, tmp1);
    k_gather_linear<<<g1(NPT), 256, 0, stream>>>(tmp1, coords, lin_w, lin_b,
                                                 (float*)d_out, NPT, 96);
}
